// LinformerSelfAttention_52639119180317
// MI455X (gfx1250) — matrix operand order repaired and checked by reading, not compiled, not run on hardware
//
#include <hip/hip_runtime.h>
#include <hip/hip_bf16.h>

// ---------------------------------------------------------------------------
// Linformer self-attention for MI455X (gfx1250), wave32, WMMA bf16 path.
// All GEMMs use v_wmma_f32_16x16x32_bf16 (D = A x B + C, fp32 accumulate).
// Attention kernel stages Kp/Vp tiles in LDS via global_load_async_to_lds_b128
// (ASYNCcnt / s_wait_asynccnt), reusing the score buffer's 32 KB.
// ---------------------------------------------------------------------------

#define BATCH   4
#define SEQ     4096
#define EMB     1024
#define HEADS   16
#define HDIM    64
#define KP      256

typedef __bf16 bf16_t;
typedef __attribute__((ext_vector_type(16))) __bf16 v16bf;
typedef __attribute__((ext_vector_type(8)))  __bf16 v8bf;
typedef __attribute__((ext_vector_type(8)))  float  v8f;

union FragBF { v16bf v; v8bf h[2]; };

// A fragment (16x32 bf16, M x K), A row-major with leading dim lda.
// Layout used here (contiguous k): lane L -> m = L%16 ; elems 0..15 = A[m][k0 + (L/16)*16 + e]
__device__ __forceinline__ v16bf load_a_frag(const bf16_t* __restrict__ A, int lda,
                                             int m0, int k0, int lane) {
  int m  = m0 + (lane & 15);
  int kk = k0 + ((lane >> 4) << 4);
  const bf16_t* p = A + (size_t)m * lda + kk;
  FragBF f;
  f.h[0] = *(const v8bf*)(p);
  f.h[1] = *(const v8bf*)(p + 8);
  return f.v;
}

// B fragment (32x16 bf16, K x N) supplied as Bt = B-transposed, row-major N x K.
// Layout used here (contiguous k, as for A): lane L -> n = L%16 ; elems 0..15 = B[k0 + (L/16)*16 + e][n]
__device__ __forceinline__ v16bf load_bt_frag(const bf16_t* __restrict__ Bt, int ldb,
                                              int n0, int k0, int lane) {
  int n  = n0 + (lane & 15);
  int kk = k0 + ((lane >> 4) << 4);
  const bf16_t* p = Bt + (size_t)n * ldb + kk;
  FragBF f;
  f.h[0] = *(const v8bf*)(p);
  f.h[1] = *(const v8bf*)(p + 8);
  return f.v;
}

__device__ __forceinline__ v8f wmma_bf16(v16bf a, v16bf b, v8f c) {
  return __builtin_amdgcn_wmma_f32_16x16x32_bf16(
      /*neg_a=*/false, a, /*neg_b=*/false, b,
      /*c_mod=*/(short)0, c, /*reuse_a=*/false, /*reuse_b=*/false);
}

// Async global->LDS 16B transfer (per-lane addresses), tracked with ASYNCcnt.
__device__ __forceinline__ void async_copy_b128(unsigned lds_byte,
                                                const bf16_t* g) {
  asm volatile("global_load_async_to_lds_b128 %0, %1, off"
               :: "v"(lds_byte), "v"((unsigned long long)(uintptr_t)g)
               : "memory");
}
__device__ __forceinline__ void wait_async0() {
  asm volatile("s_wait_asynccnt 0x0" ::: "memory");
}

// ---------------------------------------------------------------------------
// Kernel 1a: elementwise f32 -> bf16
// ---------------------------------------------------------------------------
__global__ void cvt_bf16(const float* __restrict__ src, bf16_t* __restrict__ dst, int n) {
  int i = blockIdx.x * blockDim.x + threadIdx.x;
  if (i < n) dst[i] = (bf16_t)src[i];
}

// Kernel 1b: transpose [SEQ][KP] f32 -> [KP][SEQ] bf16 (coalesced writes)
__global__ void cvt_transpose(const float* __restrict__ src, bf16_t* __restrict__ dst) {
  int i = blockIdx.x * blockDim.x + threadIdx.x;   // over KP*SEQ
  int k = i >> 12;            // SEQ = 4096
  int s = i & (SEQ - 1);
  if (k < KP) dst[i] = (bf16_t)src[s * KP + k];
}

// ---------------------------------------------------------------------------
// Kernel 2: fused QKV projection.  C[m,f] = sum_e x[m,e] * W[f,e] + b[f]
// Each wave: 16 rows x 64 cols (4 WMMA accumulators, A reused 4x).
// Q -> [B,H,S,D] bf16 ; K,V -> [B,H,D,S] bf16 (transposed for the next GEMM).
// ---------------------------------------------------------------------------
__global__ __launch_bounds__(256) void qkv_gemm(
    const bf16_t* __restrict__ xb,
    const bf16_t* __restrict__ wq, const bf16_t* __restrict__ wk,
    const bf16_t* __restrict__ wv,
    const float* __restrict__ bq, const float* __restrict__ bk,
    const float* __restrict__ bv,
    bf16_t* __restrict__ Q, bf16_t* __restrict__ Kt, bf16_t* __restrict__ Vt) {
  const int lane = threadIdx.x & 31;
  const int wave = threadIdx.x >> 5;
  const int row0 = (blockIdx.x * 8 + wave) * 16;   // over B*S = 16384
  const int col0 = blockIdx.y * 64;                // over EMB
  const int which = blockIdx.z;                    // 0=Q 1=K 2=V

  const bf16_t* W   = (which == 0) ? wq : (which == 1) ? wk : wv;
  const float* bias = (which == 0) ? bq : (which == 1) ? bk : bv;

  v8f acc[4] = {};
  for (int k0 = 0; k0 < EMB; k0 += 32) {
    if (k0 + 32 < EMB)
      __builtin_prefetch(xb + (size_t)(row0 + (lane & 15)) * EMB + k0 + 32, 0, 3);
    v16bf a = load_a_frag(xb, EMB, row0, k0, lane);
#pragma unroll
    for (int j = 0; j < 4; ++j) {
      v16bf b = load_bt_frag(W, EMB, col0 + j * 16, k0, lane);
      acc[j] = wmma_bf16(a, b, acc[j]);
    }
  }

  const int mbase = row0 + ((lane >> 4) << 3);
  const int b_ = mbase >> 12;              // batch
  const int s_ = mbase & (SEQ - 1);        // seq (8 consecutive rows share b_)
#pragma unroll
  for (int j = 0; j < 4; ++j) {
    int f = col0 + j * 16 + (lane & 15);
    float bsv = bias[f];
    int h = f >> 6, d = f & 63;
    if (which == 0) {
#pragma unroll
      for (int r = 0; r < 8; ++r) {
        Q[((((size_t)b_ * HEADS + h) * SEQ + (s_ + r)) << 6) + d] =
            (bf16_t)(acc[j][r] + bsv);
      }
    } else {
      bf16_t* dst = (which == 1) ? Kt : Vt;
      v8bf pk;
#pragma unroll
      for (int r = 0; r < 8; ++r) pk[r] = (bf16_t)(acc[j][r] + bsv);
      *(v8bf*)&dst[((((size_t)b_ * HEADS + h) * HDIM + d) << 12) + s_] = pk;
    }
  }
}

// ---------------------------------------------------------------------------
// Kernel 3: low-rank sequence projections per (b,h).
//   K_proj[k_,d] = sum_s E[s,k_] K[s,d]  -> A = Et[KPxS], Bt = Kt[DxS]
//   Vp_t [d,k_]  = sum_s F[s,k_] V[s,d]  -> A = Vt[DxS],  Bt = Ft[KPxS]
// Outputs bf16: Kp row-major [k_,d] (ld 64), Vpt row-major [d,k_] (ld 256).
// ---------------------------------------------------------------------------
__global__ __launch_bounds__(256) void lowrank_proj(
    const bf16_t* __restrict__ Et, const bf16_t* __restrict__ Ft,
    const bf16_t* __restrict__ Kt, const bf16_t* __restrict__ Vt,
    bf16_t* __restrict__ Kp, bf16_t* __restrict__ Vpt) {
  const int lane = threadIdx.x & 31;
  const int wave = threadIdx.x >> 5;
  const int bh = blockIdx.y;                       // 0..63
  int t = blockIdx.x * 8 + wave;                   // 0..127
  const bf16_t* Kt_bh = Kt + ((size_t)bh << 18);   // D*S = 262144
  const bf16_t* Vt_bh = Vt + ((size_t)bh << 18);

  v8f acc = {};
  if (t < 64) {                                    // K_proj: 16 x 4 tiles
    const int m0 = (t >> 2) * 16;                  // k_ tile
    const int n0 = (t & 3) * 16;                   // d tile
    for (int k0 = 0; k0 < SEQ; k0 += 32) {
      v16bf a = load_a_frag(Et, SEQ, m0, k0, lane);
      v16bf b = load_bt_frag(Kt_bh, SEQ, n0, k0, lane);
      acc = wmma_bf16(a, b, acc);
    }
    bf16_t* dst = Kp + ((size_t)bh << 14);         // KP*D = 16384
    int n = n0 + (lane & 15), mb = m0 + ((lane >> 4) << 3);
#pragma unroll
    for (int r = 0; r < 8; ++r) dst[(mb + r) * HDIM + n] = (bf16_t)acc[r];
  } else {                                         // Vp_t: 4 x 16 tiles
    t -= 64;
    const int m0 = (t >> 4) * 16;                  // d tile
    const int n0 = (t & 15) * 16;                  // k_ tile
    for (int k0 = 0; k0 < SEQ; k0 += 32) {
      v16bf a = load_a_frag(Vt_bh, SEQ, m0, k0, lane);
      v16bf b = load_bt_frag(Ft, SEQ, n0, k0, lane);
      acc = wmma_bf16(a, b, acc);
    }
    bf16_t* dst = Vpt + ((size_t)bh << 14);
    int n = n0 + (lane & 15), mb = m0 + ((lane >> 4) << 3);
#pragma unroll
    for (int r = 0; r < 8; ++r) dst[(mb + r) * KP + n] = (bf16_t)acc[r];
  }
}

// ---------------------------------------------------------------------------
// Kernel 4: fused attention per (b,h, 32-row s-tile).
//   Stage Kp (32 KB) in LDS (async) -> scores = Q @ Kp^T * 1/8 -> softmax
//   (fp32 probs to d_out, bf16 probs to LDS) -> stage Vp (32 KB) in the SAME
//   LDS (scores now dead) -> out = probs @ Vp -> attn_out [B,S,E] bf16.
// ---------------------------------------------------------------------------
__global__ __launch_bounds__(256) void linformer_attn(
    const bf16_t* __restrict__ Q, const bf16_t* __restrict__ Kp,
    const bf16_t* __restrict__ Vpt, bf16_t* __restrict__ AO,
    float* __restrict__ probs_out) {
  // 32 KB region, time-shared: Kp tile -> scores -> Vp tile.
  __shared__ __align__(16) union {
    float  sc[32][KP];        // fp32 scores (phase 1 out / phase 2 in)
    bf16_t mat[16384];        // Kp (phase 1) then Vpt (phase 3), flat copy
  } u;
  __shared__ __align__(16) bf16_t pb[32][KP];      // bf16 probs (16 KB)
  __shared__ float red[32][8];

  const int lane = threadIdx.x & 31;
  const int wave = threadIdx.x >> 5;
  const int bh = blockIdx.y;
  const int s0 = blockIdx.x * 32;
  const bf16_t* Qbh  = Q   + ((size_t)bh << 18);   // S*D
  const bf16_t* Kpbh = Kp  + ((size_t)bh << 14);   // KP*D  (32 KB)
  const bf16_t* Vpbh = Vpt + ((size_t)bh << 14);   // D*KP  (32 KB)

  const unsigned mat_lds = (unsigned)(uintptr_t)&u.mat[0];

  // ---- stage Kp tile into LDS: 256 threads x 8 x 16B async transfers ----
#pragma unroll
  for (int c = 0; c < 8; ++c) {
    unsigned off = (threadIdx.x + c * 256) * 16;   // bytes
    async_copy_b128(mat_lds + off, Kpbh + off / 2);
  }
  wait_async0();
  __syncthreads();

  // ---- phase 1: scores (32 x 256), wave w -> row-half w/4, 4 col tiles ----
  const int rb = wave >> 2;                        // 0..1
  {
    const int cb = (wave & 3) * 4;                 // col tile base
    v8f acc[4] = {};
    for (int k0 = 0; k0 < HDIM; k0 += 32) {
      v16bf a = load_a_frag(Qbh, HDIM, s0 + rb * 16, k0, lane);
#pragma unroll
      for (int j = 0; j < 4; ++j) {
        v16bf b = load_bt_frag(u.mat, HDIM, (cb + j) * 16, k0, lane);
        acc[j] = wmma_bf16(a, b, acc[j]);
      }
    }
    __syncthreads();   // all waves done reading Kp from u.mat
    const int lr = rb * 16 + ((lane >> 4) << 3);
#pragma unroll
    for (int j = 0; j < 4; ++j) {
      int c = (cb + j) * 16 + (lane & 15);
#pragma unroll
      for (int r = 0; r < 8; ++r) u.sc[lr + r][c] = acc[j][r] * 0.125f;
    }
  }
  __syncthreads();

  // ---- phase 2: softmax over 256 slots, 8 threads per row ----
  {
    const int row = threadIdx.x >> 3;              // 0..31
    const int tc = threadIdx.x & 7;
    const int c0 = tc * 32;
    float ev[32];
    float mx = -3.4e38f;
#pragma unroll
    for (int i = 0; i < 32; ++i) {
      float v = u.sc[row][c0 + i];
      ev[i] = v;
      mx = fmaxf(mx, v);
    }
    red[row][tc] = mx;
    __syncthreads();   // also: all reads of u.sc complete after this point
    float m = red[row][0];
#pragma unroll
    for (int i = 1; i < 8; ++i) m = fmaxf(m, red[row][i]);
    float sum = 0.f;
#pragma unroll
    for (int i = 0; i < 32; ++i) {
      float e = __expf(ev[i] - m);
      ev[i] = e;
      sum += e;
    }
    __syncthreads();
    red[row][tc] = sum;
    __syncthreads();

    // scores are dead now: start streaming Vp into the same LDS region.
#pragma unroll
    for (int c = 0; c < 8; ++c) {
      unsigned off = (threadIdx.x + c * 256) * 16;
      async_copy_b128(mat_lds + off, Vpbh + off / 2);
    }

    float tot = 0.f;
#pragma unroll
    for (int i = 0; i < 8; ++i) tot += red[row][i];
    float inv = 1.0f / tot;
    size_t gbase = (((size_t)bh << 12) + (s0 + row)) * KP + c0;
#pragma unroll
    for (int i = 0; i < 32; ++i) {
      float p = ev[i] * inv;
      probs_out[gbase + i] = p;
      pb[row][c0 + i] = (bf16_t)p;
    }
    wait_async0();
  }
  __syncthreads();

  // ---- phase 3: out(32x64) = probs(32x256) @ Vp(256x64); A,B from LDS ----
  {
    const int nb = wave & 3;
    v8f acc = {};
    for (int k0 = 0; k0 < KP; k0 += 32) {
      v16bf a = load_a_frag(&pb[0][0], KP, rb * 16, k0, lane);
      v16bf b = load_bt_frag(u.mat, KP, nb * 16, k0, lane);
      acc = wmma_bf16(a, b, acc);
    }
    const int h = bh & (HEADS - 1), b_ = bh >> 4;
    const int d = nb * 16 + (lane & 15);
    const int mb = rb * 16 + ((lane >> 4) << 3);
#pragma unroll
    for (int r = 0; r < 8; ++r) {
      int s = s0 + mb + r;
      AO[(((size_t)b_ << 12) + s) * EMB + (h << 6) + d] = (bf16_t)acc[r];
    }
  }
}

// ---------------------------------------------------------------------------
// Kernel 5: output projection, fp32 result + bias straight to d_out.
// ---------------------------------------------------------------------------
__global__ __launch_bounds__(256) void out_proj(
    const bf16_t* __restrict__ AO, const bf16_t* __restrict__ Wo,
    const float* __restrict__ bo, float* __restrict__ out) {
  const int lane = threadIdx.x & 31;
  const int wave = threadIdx.x >> 5;
  const int row0 = (blockIdx.x * 8 + wave) * 16;
  const int col0 = blockIdx.y * 64;

  v8f acc[4] = {};
  for (int k0 = 0; k0 < EMB; k0 += 32) {
    if (k0 + 32 < EMB)
      __builtin_prefetch(AO + (size_t)(row0 + (lane & 15)) * EMB + k0 + 32, 0, 3);
    v16bf a = load_a_frag(AO, EMB, row0, k0, lane);
#pragma unroll
    for (int j = 0; j < 4; ++j) {
      v16bf b = load_bt_frag(Wo, EMB, col0 + j * 16, k0, lane);
      acc[j] = wmma_bf16(a, b, acc[j]);
    }
  }
  const int mbase = row0 + ((lane >> 4) << 3);
#pragma unroll
  for (int j = 0; j < 4; ++j) {
    int f = col0 + j * 16 + (lane & 15);
    float bsv = bo[f];
#pragma unroll
    for (int r = 0; r < 8; ++r)
      out[(size_t)(mbase + r) * EMB + f] = acc[j][r] + bsv;
  }
}

// ---------------------------------------------------------------------------
extern "C" void kernel_launch(void* const* d_in, const int* in_sizes, int n_in,
                              void* d_out, int out_size, void* d_ws, size_t ws_size,
                              hipStream_t stream) {
  (void)in_sizes; (void)n_in; (void)out_size; (void)ws_size;
  const float* x  = (const float*)d_in[0];
  const float* Wq = (const float*)d_in[1];
  const float* bq = (const float*)d_in[2];
  const float* Wk = (const float*)d_in[3];
  const float* bk = (const float*)d_in[4];
  const float* Wv = (const float*)d_in[5];
  const float* bv = (const float*)d_in[6];
  const float* pE = (const float*)d_in[7];
  const float* pF = (const float*)d_in[8];
  const float* Wo = (const float*)d_in[9];
  const float* bo = (const float*)d_in[10];

  const size_t N_X = (size_t)BATCH * SEQ * EMB;   // 16,777,216
  const size_t N_W = (size_t)EMB * EMB;           // 1,048,576
  const size_t N_P = (size_t)SEQ * KP;            // 1,048,576
  const size_t N_K = (size_t)BATCH * HEADS * KP * HDIM;  // 1,048,576

  bf16_t* w    = (bf16_t*)d_ws;
  bf16_t* xb   = w; w += N_X;
  bf16_t* wqb  = w; w += N_W;
  bf16_t* wkb  = w; w += N_W;
  bf16_t* wvb  = w; w += N_W;
  bf16_t* wob  = w; w += N_W;
  bf16_t* etb  = w; w += N_P;
  bf16_t* ftb  = w; w += N_P;
  bf16_t* qb   = w; w += N_X;
  bf16_t* ktb  = w; w += N_X;
  bf16_t* vtb  = w; w += N_X;
  bf16_t* kpb  = w; w += N_K;
  bf16_t* vptb = w; w += N_K;
  bf16_t* aob  = w; w += N_X;

  float* out_f   = (float*)d_out;
  float* probs_f = out_f + N_X;   // second tuple element [B,H,S,KP]

  const int T = 256;
  cvt_bf16<<<(int)((N_X + T - 1) / T), T, 0, stream>>>(x, xb, (int)N_X);
  cvt_bf16<<<(int)((N_W + T - 1) / T), T, 0, stream>>>(Wq, wqb, (int)N_W);
  cvt_bf16<<<(int)((N_W + T - 1) / T), T, 0, stream>>>(Wk, wkb, (int)N_W);
  cvt_bf16<<<(int)((N_W + T - 1) / T), T, 0, stream>>>(Wv, wvb, (int)N_W);
  cvt_bf16<<<(int)((N_W + T - 1) / T), T, 0, stream>>>(Wo, wob, (int)N_W);
  cvt_transpose<<<(int)((N_P + T - 1) / T), T, 0, stream>>>(pE, etb);
  cvt_transpose<<<(int)((N_P + T - 1) / T), T, 0, stream>>>(pF, ftb);

  qkv_gemm<<<dim3(128, 16, 3), T, 0, stream>>>(xb, wqb, wkb, wvb, bq, bk, bv,
                                               qb, ktb, vtb);
  lowrank_proj<<<dim3(16, 64), T, 0, stream>>>(etb, ftb, ktb, vtb, kpb, vptb);
  linformer_attn<<<dim3(128, 64), T, 0, stream>>>(qb, kpb, vptb, aob, probs_f);
  out_proj<<<dim3(128, 16), T, 0, stream>>>(aob, wob, bo, out_f);
}
